// ClipLoss_58806692216888
// MI455X (gfx1250) — matrix operand order repaired and checked by reading, not compiled, not run on hardware
//
#include <hip/hip_runtime.h>
#include <hip/hip_bf16.h>

#define CLIP_N 16384
#define CLIP_D 512
#define TN 4                         // column tiles (of 16) per group
#define WAVES 8                      // waves per workgroup -> 128 rows
#define NRB   (CLIP_N / (16 * WAVES))  // 128 row blocks
#define JSPLIT 2                     // column-sweep split for occupancy
#define NEG_BIG (-3.0e38f)

typedef __attribute__((ext_vector_type(16))) __bf16 bf16x16;
typedef __attribute__((ext_vector_type(8)))  __bf16 bf16x8;
typedef __attribute__((ext_vector_type(8)))  float  f32x8;

// ---------------------------------------------------------------------------
// A-operand fragment: lane holds row (lane&15); elements [0..7] = K kb..kb+7,
// elements [8..15] = K kb+16..kb+23 (kb = 0 lanes 0-15, 8 lanes 16-31).
// ---------------------------------------------------------------------------
__device__ __forceinline__ bf16x16 load_a_frag(const __bf16* base) {
    bf16x8 p0 = *(const bf16x8*)(base);
    bf16x8 p1 = *(const bf16x8*)(base + 16);
    return __builtin_shufflevector(p0, p1, 0, 1, 2, 3, 4, 5, 6, 7,
                                            8, 9, 10, 11, 12, 13, 14, 15);
}

// ---------------------------------------------------------------------------
// Split f32 -> bf16 hi + bf16 lo residual
// ---------------------------------------------------------------------------
__global__ void split_bf16_kernel(const float* __restrict__ x,
                                  __bf16* __restrict__ hi,
                                  __bf16* __restrict__ lo, int n) {
    int i = blockIdx.x * blockDim.x + threadIdx.x;
    if (i < n) {
        float v  = x[i];
        __bf16 h = (__bf16)v;
        hi[i] = h;
        lo[i] = (__bf16)(v - (float)h);
    }
}

// ---------------------------------------------------------------------------
// Single-pass fused GEMM producing BOTH row-LSE partials and column-LSE
// partials (base-2 max/sum pairs).
//   logits2[i][j] = (scale/ln2) * A[i].B[j]  via 3x bf16 WMMA split product
// Each workgroup: 128 rows (8 waves x 16), sweeps half the columns (JSPLIT).
//   rowM/rowS: [row][JSPLIT]   (partials over this workgroup's column half)
//   colM/colS: [col][NRB]      (partials over this workgroup's 128 rows)
// ---------------------------------------------------------------------------
__global__ __launch_bounds__(WAVES * 32, 1)
void clip_fused_kernel(const __bf16* __restrict__ Ah,
                       const __bf16* __restrict__ Al,
                       const __bf16* __restrict__ Bh,
                       const __bf16* __restrict__ Bl,
                       const float* __restrict__ scale_ptr,
                       float* __restrict__ rowM, float* __restrict__ rowS,
                       float* __restrict__ colM, float* __restrict__ colS) {
    const int lane    = threadIdx.x & 31;
    const int wave    = threadIdx.x >> 5;
    const int l15     = lane & 15;
    const int halfsel = lane >> 4;                 // 0: lanes 0-15, 1: 16-31
    const int rb      = blockIdx.x >> 1;           // row block, 0..NRB-1
    const int jh      = blockIdx.x & 1;            // column half, 0..JSPLIT-1
    const int rowbase = (rb * WAVES + wave) * 16;

    const float scale2 = scale_ptr[0] * 1.4426950408889634f;  // scale/ln2

    const __bf16* aRowH = Ah + (size_t)(rowbase + l15) * CLIP_D + (halfsel ? 8 : 0);
    const __bf16* aRowL = Al + (size_t)(rowbase + l15) * CLIP_D + (halfsel ? 8 : 0);
    const int kbB = halfsel ? 8 : 0;               // B-operand K sub-offset

    // Hoist ALL A fragments for K=512 into registers (16 chunks x hi/lo).
    bf16x16 aH[CLIP_D / 32], aL[CLIP_D / 32];
#pragma unroll
    for (int k = 0; k < CLIP_D / 32; ++k) {
        aH[k] = load_a_frag(aRowH + k * 32);
        aL[k] = load_a_frag(aRowL + k * 32);
    }

    // Row online-LSE state (base-2): VGPR r <-> row rowbase + r + 8*halfsel
    float m[8], s[8];
#pragma unroll
    for (int r = 0; r < 8; ++r) { m[r] = NEG_BIG; s[r] = 0.0f; }

    __shared__ float lds_m[WAVES][TN][16];
    __shared__ float lds_s[WAVES][TN][16];

    const int tiles_per_half = (CLIP_N / 16) / JSPLIT;
    const int tile_lo = jh * tiles_per_half;
    const int tile_hi = tile_lo + tiles_per_half;

    for (int jt0 = tile_lo; jt0 < tile_hi; jt0 += TN) {
        f32x8 c[TN];
#pragma unroll
        for (int t = 0; t < TN; ++t)
#pragma unroll
            for (int r = 0; r < 8; ++r) c[t][r] = 0.0f;

        const __bf16* bRowH[TN];
        const __bf16* bRowL[TN];
#pragma unroll
        for (int t = 0; t < TN; ++t) {
            size_t col = (size_t)(jt0 + t) * 16 + l15;
            bRowH[t] = Bh + col * CLIP_D + kbB;
            bRowL[t] = Bl + col * CLIP_D + kbB;
        }

        // ---- WMMA core: (ah+al)*(bh+bl) ~= ah*bh + ah*bl + al*bh ----
#pragma unroll
        for (int k = 0; k < CLIP_D / 32; ++k) {
#pragma unroll
            for (int t = 0; t < TN; ++t) {
                bf16x16 b_h = load_a_frag(bRowH[t] + k * 32);
                bf16x16 b_l = load_a_frag(bRowL[t] + k * 32);
                c[t] = __builtin_amdgcn_wmma_f32_16x16x32_bf16(
                           false, aH[k], false, b_h, (short)0, c[t], false, false);
                c[t] = __builtin_amdgcn_wmma_f32_16x16x32_bf16(
                           false, aH[k], false, b_l, (short)0, c[t], false, false);
                c[t] = __builtin_amdgcn_wmma_f32_16x16x32_bf16(
                           false, aL[k], false, b_h, (short)0, c[t], false, false);
            }
        }

        // scale into base-2 logits (in place)
#pragma unroll
        for (int t = 0; t < TN; ++t)
#pragma unroll
            for (int r = 0; r < 8; ++r) c[t][r] *= scale2;

        // ---- row online LSE update (group max first: 5 exps per row) ----
#pragma unroll
        for (int r = 0; r < 8; ++r) {
            float g  = fmaxf(fmaxf(c[0][r], c[1][r]), fmaxf(c[2][r], c[3][r]));
            float mo = m[r];
            float mn = fmaxf(mo, g);
            float acc = __builtin_amdgcn_exp2f(c[0][r] - mn)
                      + __builtin_amdgcn_exp2f(c[1][r] - mn)
                      + __builtin_amdgcn_exp2f(c[2][r] - mn)
                      + __builtin_amdgcn_exp2f(c[3][r] - mn);
            s[r] = s[r] * __builtin_amdgcn_exp2f(mo - mn) + acc;
            m[r] = mn;
        }

        // ---- column partials: per lane = one column over 8 rows ----
#pragma unroll
        for (int t = 0; t < TN; ++t) {
            float cm = c[t][0];
#pragma unroll
            for (int r = 1; r < 8; ++r) cm = fmaxf(cm, c[t][r]);
            float cs = 0.0f;
#pragma unroll
            for (int r = 0; r < 8; ++r)
                cs += __builtin_amdgcn_exp2f(c[t][r] - cm);
            // merge the two lane halves (rows 0-7 with rows 8-15)
            float om = __shfl_xor(cm, 16, 32);
            float os = __shfl_xor(cs, 16, 32);
            float nm = fmaxf(cm, om);
            cs = cs * __builtin_amdgcn_exp2f(cm - nm)
               + os * __builtin_amdgcn_exp2f(om - nm);
            cm = nm;
            if (halfsel == 0) {
                lds_m[wave][t][l15] = cm;
                lds_s[wave][t][l15] = cs;
            }
        }
        __syncthreads();
        // reduce 8 waves -> workgroup column partial; write [col][NRB]
        if (threadIdx.x < TN * 16) {
            int t  = threadIdx.x >> 4;
            int cl = threadIdx.x & 15;
            float M = NEG_BIG, S = 0.0f;
#pragma unroll
            for (int w = 0; w < WAVES; ++w) {
                float mw = lds_m[w][t][cl];
                float sw = lds_s[w][t][cl];
                float Mn = fmaxf(M, mw);
                S = S * __builtin_amdgcn_exp2f(M - Mn)
                  + sw * __builtin_amdgcn_exp2f(mw - Mn);
                M = Mn;
            }
            int col = jt0 * 16 + threadIdx.x;
            colM[(size_t)col * NRB + rb] = M;
            colS[(size_t)col * NRB + rb] = S;
        }
        __syncthreads();
    }

    // ---- row partial: merge 16 lanes per half, write [row][JSPLIT] ----
#pragma unroll
    for (int off = 1; off < 16; off <<= 1) {
#pragma unroll
        for (int r = 0; r < 8; ++r) {
            float mo = __shfl_xor(m[r], off, 32);
            float so = __shfl_xor(s[r], off, 32);
            float mn = fmaxf(m[r], mo);
            s[r] = s[r] * __builtin_amdgcn_exp2f(m[r] - mn)
                 + so * __builtin_amdgcn_exp2f(mo - mn);
            m[r] = mn;
        }
    }
    if (l15 == 0) {
#pragma unroll
        for (int r = 0; r < 8; ++r) {
            int row = rowbase + r + (halfsel << 3);
            rowM[(size_t)row * JSPLIT + jh] = m[r];
            rowS[(size_t)row * JSPLIT + jh] = s[r];
        }
    }
}

// ---------------------------------------------------------------------------
// Merge nPart (max,sum) partials per output (layout [out][nPart]) -> natural
// logsumexp. One wave per output.
// ---------------------------------------------------------------------------
__global__ void lse_merge_kernel(const float* __restrict__ pm,
                                 const float* __restrict__ ps,
                                 int nPart, float* __restrict__ out) {
    int idx  = blockIdx.x * (blockDim.x >> 5) + (threadIdx.x >> 5);
    int lane = threadIdx.x & 31;
    float M = NEG_BIG, S = 0.0f;
    for (int p = lane; p < nPart; p += 32) {
        float mw = pm[(size_t)idx * nPart + p];
        float sw = ps[(size_t)idx * nPart + p];
        float Mn = fmaxf(M, mw);
        S = S * __builtin_amdgcn_exp2f(M - Mn)
          + sw * __builtin_amdgcn_exp2f(mw - Mn);
        M = Mn;
    }
#pragma unroll
    for (int off = 16; off > 0; off >>= 1) {
        float mo = __shfl_xor(M, off, 32);
        float so = __shfl_xor(S, off, 32);
        float Mn = fmaxf(M, mo);
        S = S * __builtin_amdgcn_exp2f(M - Mn)
          + so * __builtin_amdgcn_exp2f(mo - Mn);
        M = Mn;
    }
    if (lane == 0)
        out[idx] = 0.6931471805599453f * (M + __builtin_amdgcn_logf(S));
}

// ---------------------------------------------------------------------------
// Exact fp32 diagonal: diag[i] = scale * dot(image[i], text[i]); one wave/row
// ---------------------------------------------------------------------------
__global__ void diag_kernel(const float* __restrict__ img,
                            const float* __restrict__ txt,
                            const float* __restrict__ scale_ptr,
                            float* __restrict__ diag) {
    int row  = blockIdx.x * (blockDim.x >> 5) + (threadIdx.x >> 5);
    int lane = threadIdx.x & 31;
    if (row >= CLIP_N) return;
    const float* a = img + (size_t)row * CLIP_D;
    const float* b = txt + (size_t)row * CLIP_D;
    float acc = 0.0f;
    for (int k = lane; k < CLIP_D; k += 32) acc += a[k] * b[k];
#pragma unroll
    for (int off = 16; off > 0; off >>= 1) acc += __shfl_xor(acc, off, 32);
    if (lane == 0) diag[row] = acc * scale_ptr[0];
}

// ---------------------------------------------------------------------------
// loss = ( sum(lse_rows) + sum(lse_cols) - 2*sum(diag) ) / (2N)
// ---------------------------------------------------------------------------
__global__ void final_reduce_kernel(const float* __restrict__ lse_r,
                                    const float* __restrict__ lse_c,
                                    const float* __restrict__ diag,
                                    float* __restrict__ out) {
    __shared__ float red[256];
    float acc = 0.0f;
    for (int i = threadIdx.x; i < CLIP_N; i += 256)
        acc += lse_r[i] + lse_c[i] - 2.0f * diag[i];
    red[threadIdx.x] = acc;
    __syncthreads();
    for (int stride = 128; stride > 0; stride >>= 1) {
        if (threadIdx.x < stride) red[threadIdx.x] += red[threadIdx.x + stride];
        __syncthreads();
    }
    if (threadIdx.x == 0) out[0] = red[0] / (2.0f * (float)CLIP_N);
}

// ---------------------------------------------------------------------------
extern "C" void kernel_launch(void* const* d_in, const int* in_sizes, int n_in,
                              void* d_out, int out_size, void* d_ws, size_t ws_size,
                              hipStream_t stream) {
    const float* d_img   = (const float*)d_in[0];   // [N, D] f32
    const float* d_txt   = (const float*)d_in[1];   // [N, D] f32
    const float* d_scale = (const float*)d_in[2];   // scalar f32
    float* out = (float*)d_out;

    const size_t nd = (size_t)CLIP_N * CLIP_D;

    // workspace layout
    char* ws = (char*)d_ws;
    size_t off = 0;
    __bf16* imgH = (__bf16*)(ws + off); off += nd * sizeof(__bf16);
    __bf16* imgL = (__bf16*)(ws + off); off += nd * sizeof(__bf16);
    __bf16* txtH = (__bf16*)(ws + off); off += nd * sizeof(__bf16);
    __bf16* txtL = (__bf16*)(ws + off); off += nd * sizeof(__bf16);
    float* colM  = (float*)(ws + off);  off += (size_t)CLIP_N * NRB * sizeof(float);
    float* colS  = (float*)(ws + off);  off += (size_t)CLIP_N * NRB * sizeof(float);
    float* rowM  = (float*)(ws + off);  off += (size_t)CLIP_N * JSPLIT * sizeof(float);
    float* rowS  = (float*)(ws + off);  off += (size_t)CLIP_N * JSPLIT * sizeof(float);
    float* lse_r = (float*)(ws + off);  off += (size_t)CLIP_N * sizeof(float);
    float* lse_c = (float*)(ws + off);  off += (size_t)CLIP_N * sizeof(float);
    float* diag  = (float*)(ws + off);  off += (size_t)CLIP_N * sizeof(float);

    // 1) split inputs into bf16 hi/lo
    {
        int threads = 256;
        int blocks  = (int)((nd + threads - 1) / threads);
        split_bf16_kernel<<<blocks, threads, 0, stream>>>(d_img, imgH, imgL, (int)nd);
        split_bf16_kernel<<<blocks, threads, 0, stream>>>(d_txt, txtH, txtL, (int)nd);
    }

    // 2) single fused GEMM pass -> row partials + column partials
    clip_fused_kernel<<<dim3(NRB * JSPLIT), dim3(WAVES * 32), 0, stream>>>(
        imgH, imgL, txtH, txtL, d_scale, rowM, rowS, colM, colS);

    // 3) merge partials into lse arrays (one wave per output)
    lse_merge_kernel<<<dim3(CLIP_N / 8), dim3(256), 0, stream>>>(rowM, rowS,
                                                                 JSPLIT, lse_r);
    lse_merge_kernel<<<dim3(CLIP_N / 8), dim3(256), 0, stream>>>(colM, colS,
                                                                 NRB, lse_c);

    // 4) exact diagonal in fp32
    diag_kernel<<<dim3(CLIP_N / 8), dim3(256), 0, stream>>>(d_img, d_txt,
                                                            d_scale, diag);

    // 5) final scalar loss
    final_reduce_kernel<<<1, 256, 0, stream>>>(lse_r, lse_c, diag, out);
}
